// GatedDeltaNet_32023276159039
// MI455X (gfx1250) — matrix operand order repaired and checked by reading, not compiled, not run on hardware
//
#include <hip/hip_runtime.h>
#include <hip/hip_bf16.h>
#include <math.h>

// ---------------------------------------------------------------------------
// Problem constants (GatedDeltaNet reference)
// ---------------------------------------------------------------------------
constexpr int cB   = 2;
constexpr int cT   = 1024;
constexpr int cHID = 2048;
constexpr int cNKH = 8, cNVH = 16, cDK = 64, cDV = 128, cKS = 4;
constexpr int cKD  = cNKH * cDK;            // 512
constexpr int cVD  = cNVH * cDV;            // 2048
constexpr int cCD  = 2 * cKD + cVD;         // 3072
constexpr int cIPD = cCD + cVD + 2 * cNVH;  // 5152
constexpr int cM   = cB * cT;               // 2048 rows for both GEMMs
constexpr float cSCALE = 0.125f;            // DK^-0.5
constexpr float cEPS   = 1e-6f;

// ---------------------------------------------------------------------------
// WMMA types (CDNA5 gfx1250: wave32, v_wmma_f32_16x16x32_bf16)
// ---------------------------------------------------------------------------
typedef __attribute__((ext_vector_type(16))) __bf16 v16bf;
typedef __attribute__((ext_vector_type(8)))  float  v8f;
typedef __attribute__((ext_vector_type(4)))  int    v4i;

union BfFrag {
    v16bf bf;
    v4i   q[2];
};

// ---------------------------------------------------------------------------
// CDNA5 async global->LDS staging (ASYNCcnt-tracked); guarded fallback.
// Probe round 2 revealed the builtin's signature: arg0 is v4i addrspace(1)*
// (clang prints addrspace(1) as `__device__`), arg1 the LDS-side pointer,
// then (int offset, int cpol).
// ---------------------------------------------------------------------------
#if defined(__has_builtin)
#  if __has_builtin(__builtin_amdgcn_global_load_async_to_lds_b128)
#    define GDN_HAVE_ASYNC_LDS 1
#  endif
#  if __has_builtin(__builtin_amdgcn_s_wait_asynccnt)
#    define GDN_HAVE_WAIT_ASYNC 1
#  endif
#endif

typedef __attribute__((address_space(1))) v4i as1_v4i;
typedef __attribute__((address_space(3))) v4i as3_v4i;

__device__ __forceinline__ void async_copy16(const __bf16* g, __bf16* l) {
#if defined(GDN_HAVE_ASYNC_LDS)
    __builtin_amdgcn_global_load_async_to_lds_b128(
        (as1_v4i*)g, (as3_v4i*)l, 0, 0);
#else
    *(v4i*)l = *(const v4i*)g;   // global->reg->LDS; fenced by __syncthreads
#endif
}

__device__ __forceinline__ void wait_async0() {
#if defined(GDN_HAVE_ASYNC_LDS)
#  if defined(GDN_HAVE_WAIT_ASYNC)
    __builtin_amdgcn_s_wait_asynccnt(0);
#  else
    asm volatile("s_wait_asynccnt 0x0" ::: "memory");
#  endif
#endif
}

// ---------------------------------------------------------------------------
// f32 -> bf16 conversion (RNE)
// ---------------------------------------------------------------------------
__global__ void cvt_f32_to_bf16(const float* __restrict__ in,
                                __bf16* __restrict__ out, long long n) {
    long long i = (long long)blockIdx.x * blockDim.x + threadIdx.x;
    if (i < n) out[i] = (__bf16)in[i];
}

// ---------------------------------------------------------------------------
// C[M,N] = A[M,K] * W[N,K]^T   (bf16 in, f32 accumulate, WMMA 16x16x32)
//
// Block: 256 threads = 8 waves; wave w owns rows [m0+16w, +16); all waves
// share one 64-column B tile per K-step, staged into LDS (double-buffered,
// async copy) since fragments are identical across waves.  A fragments are
// register double-buffered.  Pipeline per iteration:
//    s_wait_asynccnt 0 ; barrier ; issue async(next tile) ; compute(current)
// so the async copy for tile k+1 overlaps all 4 WMMAs of tile k.
//
// Fragment addressing (ISA VGPR-layout tables):
//  A 16x32 bf16 : row = lane&15;  K = kb + (lane>>4)*8 (+16) .. +7
//  B 32x16 bf16 : col = lane&15;  K = kb + (lane>>4)*8 (+16) .. +7  (from [N,K])
//  C/D 16x16 f32: col = lane&15;  VGPR r = row r + 8*(lane>>4)
// ---------------------------------------------------------------------------
__global__ __launch_bounds__(256)
void wmma_gemm_bf16_nt(const __bf16* __restrict__ A,
                       const __bf16* __restrict__ W,
                       float* __restrict__ C,
                       int M, int N, int K) {
    constexpr int LSTR = 40;               // padded row stride (80B; 16B-aligned)
    __shared__ __bf16 smem[2][64 * LSTR];  // 2 x 5 KB B tiles

    const int tid  = threadIdx.x;
    const int lane = tid & 31;
    const int wave = tid >> 5;
    const int half = lane >> 4;
    const int l16  = lane & 15;

    const int m0 = blockIdx.x * 128 + wave * 16;
    const int n0 = blockIdx.y * 64;

    // cooperative B staging: thread t copies 16B: n = t>>2, k0 = (t&3)*8
    const int bn  = tid >> 2;
    const int bk0 = (tid & 3) * 8;
    int nc = n0 + bn;
    nc = (nc < N) ? nc : (N - 1);          // clamp at ragged N edge
    const __bf16* gB  = W + (size_t)nc * K + bk0;
    __bf16* lB0 = &smem[0][bn * LSTR + bk0];
    __bf16* lB1 = &smem[1][bn * LSTR + bk0];

    const __bf16* arow = A + (size_t)(m0 + l16) * K;

    v8f acc[4] = {};
    BfFrag a_cur, a_nxt;
    a_cur.q[0] = *(const v4i*)(arow + half * 8);
    a_cur.q[1] = *(const v4i*)(arow + 16 + half * 8);
    a_nxt = a_cur;

    async_copy16(gB, lB0);                 // stage tile kb=0

    for (int kb = 0; kb < K; kb += 32) {
        const int cur = (kb >> 5) & 1;
        wait_async0();                     // own slice of tile[cur] landed
        __syncthreads();                   // everyone's slice landed;
                                           // tile[cur^1] reads finished

        if (kb + 32 < K) {                 // overlap next-tile copy w/ compute
            async_copy16(gB + kb + 32, cur ? lB0 : lB1);
            a_nxt.q[0] = *(const v4i*)(arow + kb + 32 + half * 8);
            a_nxt.q[1] = *(const v4i*)(arow + kb + 48 + half * 8);
            __builtin_prefetch(arow + kb + 64, 0, 1);   // global_prefetch_b8
        }

        const __bf16* bbuf = smem[cur];
        BfFrag b[4];
#pragma unroll
        for (int j = 0; j < 4; ++j) {
            const __bf16* bp = bbuf + (j * 16 + l16) * LSTR + half * 8;
            b[j].q[0] = *(const v4i*)(bp);
            b[j].q[1] = *(const v4i*)(bp + 16);
        }
#pragma unroll
        for (int j = 0; j < 4; ++j)
            acc[j] = __builtin_amdgcn_wmma_f32_16x16x32_bf16(
                false, a_cur.bf, false, b[j].bf, (short)0, acc[j], false, false);

        a_cur = a_nxt;
    }

#pragma unroll
    for (int j = 0; j < 4; ++j) {
        int n = n0 + j * 16 + l16;
        if (n < N) {
#pragma unroll
            for (int r = 0; r < 8; ++r) {
                int m = m0 + half * 8 + r;
                C[(size_t)m * N + n] = acc[j][r];
            }
        }
    }
}

// ---------------------------------------------------------------------------
// Depthwise causal conv (K=4) + SiLU: qkv[b,t,c] = silu(sum_k w[c,k]*mix[b,t-3+k,c])
// ---------------------------------------------------------------------------
__global__ void conv_silu_kernel(const float* __restrict__ proj,
                                 const float* __restrict__ conv_w,
                                 float* __restrict__ qkv) {
    long long i = (long long)blockIdx.x * blockDim.x + threadIdx.x;
    if (i >= (long long)cM * cCD) return;
    int c = (int)(i % cCD);
    long long bt = i / cCD;
    int t = (int)(bt % cT);
    float acc = 0.f;
#pragma unroll
    for (int k = 0; k < cKS; ++k) {
        int tt = t - (cKS - 1) + k;
        if (tt >= 0)
            acc = fmaf(conv_w[c * cKS + k],
                       proj[(bt + (tt - t)) * (long long)cIPD + c], acc);
    }
    qkv[i] = acc / (1.f + __expf(-acc));   // SiLU
}

// ---------------------------------------------------------------------------
// Per-(b,t): L2-normalize q,k per head; beta = sigmoid(b),
// g = -exp(A_log)*softplus(a+dt_bias).  8 waves; wave w owns head w (2 el/lane).
// ---------------------------------------------------------------------------
__global__ __launch_bounds__(256)
void prep_kernel(const float* __restrict__ qkv, const float* __restrict__ proj,
                 const float* __restrict__ dt_bias, const float* __restrict__ A_log,
                 float* __restrict__ qn, float* __restrict__ kn,
                 float* __restrict__ gout, float* __restrict__ betaout) {
    long long bt = blockIdx.x;
    int tid  = threadIdx.x;
    int head = tid >> 5;
    int lane = tid & 31;

    __shared__ float sq[cNKH], sk[cNKH];

    const float* qbase = qkv + bt * cCD + head * cDK;
    const float* kbase = qbase + cKD;
    float q0 = qbase[lane], q1 = qbase[lane + 32];
    float k0 = kbase[lane], k1 = kbase[lane + 32];
    float pq = q0 * q0 + q1 * q1;
    float pk = k0 * k0 + k1 * k1;
#pragma unroll
    for (int off = 16; off > 0; off >>= 1) {
        pq += __shfl_xor(pq, off, 32);
        pk += __shfl_xor(pk, off, 32);
    }
    if (lane == 0) { sq[head] = pq; sk[head] = pk; }
    __syncthreads();

    float invq = 1.f / fmaxf(sqrtf(sq[head]), 1e-12f);
    float invk = 1.f / fmaxf(sqrtf(sk[head]), 1e-12f);
    float* qd = qn + bt * cKD + head * cDK;
    float* kd = kn + bt * cKD + head * cDK;
    qd[lane] = q0 * invq; qd[lane + 32] = q1 * invq;
    kd[lane] = k0 * invk; kd[lane + 32] = k1 * invk;

    if (tid < cNVH) {
        float bv = proj[bt * cIPD + cCD + cVD + tid];
        float av = proj[bt * cIPD + cCD + cVD + cNVH + tid];
        float x  = av + dt_bias[tid];
        float sp = (x > 20.f) ? x : log1pf(__expf(x));
        gout[bt * cNVH + tid]    = -__expf(A_log[tid]) * sp;
        betaout[bt * cNVH + tid] = 1.f / (1.f + __expf(-bv));
    }
}

// ---------------------------------------------------------------------------
// Sequential delta-rule scan: state columns are independent; one block of
// 128 threads per (b,h); thread v keeps state[:,v] (64 f32) in registers,
// k_t / q_t staged through LDS each step.
// ---------------------------------------------------------------------------
__global__ __launch_bounds__(128)
void scan_kernel(const float* __restrict__ qn, const float* __restrict__ kn,
                 const float* __restrict__ qkv, const float* __restrict__ g,
                 const float* __restrict__ beta, float* __restrict__ o) {
    const int h  = blockIdx.x % cNVH;
    const int b  = blockIdx.x / cNVH;
    const int hk = h / (cNVH / cNKH);
    const int v  = threadIdx.x;

    float s[cDK];
#pragma unroll
    for (int i = 0; i < cDK; ++i) s[i] = 0.f;

    __shared__ float kq[2 * cDK];

    for (int t = 0; t < cT; ++t) {
        long long bt = (long long)b * cT + t;
        if (v < cDK) kq[v] = kn[bt * cKD + hk * cDK + v];
        else         kq[v] = qn[bt * cKD + hk * cDK + (v - cDK)];
        __syncthreads();

        float eg = __expf(g[bt * cNVH + h]);
        float bw = beta[bt * cNVH + h];
        float vt = qkv[bt * cCD + 2 * cKD + h * cDV + v];

        float sk = 0.f;
#pragma unroll
        for (int i = 0; i < cDK; ++i) { s[i] *= eg; sk = fmaf(s[i], kq[i], sk); }
        float delta = bw * (vt - sk);
        float out = 0.f;
#pragma unroll
        for (int i = 0; i < cDK; ++i) {
            s[i] = fmaf(kq[i], delta, s[i]);
            out  = fmaf(s[i], kq[cDK + i], out);
        }
        o[bt * cVD + h * cDV + v] = out * cSCALE;
        __syncthreads();
    }
}

// ---------------------------------------------------------------------------
// RMS-norm over DV + SiLU(z) gate; emit bf16 directly for the out_proj GEMM.
// ---------------------------------------------------------------------------
__global__ __launch_bounds__(128)
void norm_gate_kernel(const float* __restrict__ o, const float* __restrict__ proj,
                      const float* __restrict__ norm_w, __bf16* __restrict__ gated) {
    int h = blockIdx.x % cNVH;
    long long bt = blockIdx.x / cNVH;
    int v = threadIdx.x;
    int lane = v & 31, wave = v >> 5;

    float ov = o[bt * cVD + h * cDV + v];
    float p = ov * ov;
#pragma unroll
    for (int off = 16; off > 0; off >>= 1) p += __shfl_xor(p, off, 32);
    __shared__ float red[4];
    if (lane == 0) red[wave] = p;
    __syncthreads();
    float sum = red[0] + red[1] + red[2] + red[3];
    float inv = rsqrtf(sum * (1.f / cDV) + cEPS);

    float z  = proj[bt * cIPD + cCD + (long long)h * cDV + v];
    float sz = z / (1.f + __expf(-z));
    gated[bt * cVD + h * cDV + v] = (__bf16)(norm_w[v] * ov * inv * sz);
}

// ---------------------------------------------------------------------------
// Launch
// ---------------------------------------------------------------------------
extern "C" void kernel_launch(void* const* d_in, const int* in_sizes, int n_in,
                              void* d_out, int out_size, void* d_ws, size_t ws_size,
                              hipStream_t stream) {
    (void)in_sizes; (void)n_in; (void)out_size; (void)ws_size;

    const float* x       = (const float*)d_in[0];
    /* d_in[1] = input_pos (unused: positions are 0..T-1) */
    const float* in_w    = (const float*)d_in[2];
    const float* conv_w  = (const float*)d_in[3];
    const float* dt_bias = (const float*)d_in[4];
    const float* A_log   = (const float*)d_in[5];
    const float* norm_w  = (const float*)d_in[6];
    const float* out_w   = (const float*)d_in[7];
    float* out = (float*)d_out;

    char* ws = (char*)d_ws;
    size_t off = 0;
    auto alloc = [&](size_t bytes) {
        size_t cur = off;
        off += (bytes + 255) & ~(size_t)255;
        return cur;
    };
    float*  proj = (float*)(ws + alloc((size_t)cM * cIPD * 4));
    __bf16* xbf  = (__bf16*)(ws + alloc((size_t)cM * cHID * 2));    // -> gated
    __bf16* w1bf = (__bf16*)(ws + alloc((size_t)cIPD * cHID * 2));  // -> o
    float*  qkv  = (float*)(ws + alloc((size_t)cM * cCD * 4));
    float*  qn   = (float*)(ws + alloc((size_t)cM * cKD * 4));      // -> w2bf
    float*  kn   = (float*)(ws + alloc((size_t)cM * cKD * 4));
    float*  gbuf = (float*)(ws + alloc((size_t)cM * cNVH * 4));
    float*  bbuf = (float*)(ws + alloc((size_t)cM * cNVH * 4));
    float*  obuf  = (float*)w1bf;
    __bf16* gated = (__bf16*)xbf;
    __bf16* w2bf  = (__bf16*)qn;

    // 1) bf16 copies of x and in_proj_w
    long long nx = (long long)cM * cHID;
    cvt_f32_to_bf16<<<(unsigned)((nx + 255) / 256), 256, 0, stream>>>(x, xbf, nx);
    long long nw1 = (long long)cIPD * cHID;
    cvt_f32_to_bf16<<<(unsigned)((nw1 + 255) / 256), 256, 0, stream>>>(in_w, w1bf, nw1);

    // 2) proj = x @ in_proj_w^T
    dim3 g1(cM / 128, (cIPD + 63) / 64);
    wmma_gemm_bf16_nt<<<g1, 256, 0, stream>>>(xbf, w1bf, proj, cM, cIPD, cHID);

    // 3) causal depthwise conv + SiLU
    long long nc = (long long)cM * cCD;
    conv_silu_kernel<<<(unsigned)((nc + 255) / 256), 256, 0, stream>>>(proj, conv_w, qkv);

    // 4) l2norm q/k + gate scalars
    prep_kernel<<<cM, 256, 0, stream>>>(qkv, proj, dt_bias, A_log, qn, kn, gbuf, bbuf);

    // 5) delta-rule scan
    scan_kernel<<<cB * cNVH, 128, 0, stream>>>(qn, kn, qkv, gbuf, bbuf, obuf);

    // 6) RMS-norm + SiLU(z) gate -> bf16
    norm_gate_kernel<<<cM * cNVH, 128, 0, stream>>>(obuf, proj, norm_w, gated);

    // 7) out = gated @ out_proj_w^T
    long long nw2 = (long long)cHID * cVD;
    cvt_f32_to_bf16<<<(unsigned)((nw2 + 255) / 256), 256, 0, stream>>>(out_w, w2bf, nw2);
    dim3 g2(cM / 128, cHID / 64);
    wmma_gemm_bf16_nt<<<g2, 256, 0, stream>>>(gated, w2bf, out, cM, cHID, cVD);
}
